// GatedDeltaBlock_65403761983523
// MI455X (gfx1250) — compile-verified, matrix operand order checked by reading, not run on hardware
//
#include <hip/hip_runtime.h>
#include <hip/hip_bf16.h>

typedef __attribute__((ext_vector_type(16))) __bf16 v16bf;
typedef __attribute__((ext_vector_type(8)))  float  v8f;

#define B_  4
#define T_  2048
#define D_  1024
#define H_  8
#define HD_ 128
#define FOUR_D_ 4096
static constexpr float RES_SCALE_ = 0.5f; // 1/sqrt(2*2)

// --------------------------------------------------------------------------
// gfx1250 async global->LDS copy (ASYNCcnt path). The builtin's parameters
// are pointers to <4 x i32> in global (AS1) / LDS (AS3) address spaces.
// Guarded so the build is safe on toolchains that do not declare it.
// --------------------------------------------------------------------------
#if __has_builtin(__builtin_amdgcn_global_load_async_to_lds_b128)
#define ASYNC_LDS 1
#warning "CDNA5 probe: ASYNC_LDS ENABLED (__builtin_amdgcn_global_load_async_to_lds_b128 declared)"
typedef int v4i_raw __attribute__((vector_size(16)));
typedef __attribute__((address_space(1))) v4i_raw* gptr_b128;
typedef __attribute__((address_space(3))) v4i_raw* lptr_b128;
__device__ __forceinline__ void async_cp16(const float* g, float* l) {
  __builtin_amdgcn_global_load_async_to_lds_b128((gptr_b128)g, (lptr_b128)l, 0, 0);
}
__device__ __forceinline__ void async_wait0() {
#if __has_builtin(__builtin_amdgcn_s_wait_asynccnt)
  __builtin_amdgcn_s_wait_asynccnt(0);
#else
  asm volatile("s_wait_asynccnt 0x0" ::: "memory");
#endif
}
#else
#define ASYNC_LDS 0
#warning "CDNA5 probe: ASYNC_LDS DISABLED (builtin not declared; plain-load fallback)"
#endif

#if __has_builtin(__builtin_amdgcn_s_wait_asynccnt)
#warning "CDNA5 probe: s_wait_asynccnt builtin declared"
#endif

// ---------------------------------------------------------------------------
// Transpose + f32->bf16 convert:  WT[n*K+k] = (bf16) W[k*N+n]
// ---------------------------------------------------------------------------
__global__ void k_transpose_cvt(const float* __restrict__ W, __bf16* __restrict__ WT,
                                int K, int N) {
  __shared__ float tile[32][33];
  int k0 = blockIdx.y * 32, n0 = blockIdx.x * 32;
  int tx = threadIdx.x, ty = threadIdx.y;            // block (32, 8)
  #pragma unroll
  for (int i = 0; i < 32; i += 8)
    tile[ty + i][tx] = W[(size_t)(k0 + ty + i) * N + (n0 + tx)];
  __syncthreads();
  #pragma unroll
  for (int i = 0; i < 32; i += 8)
    WT[(size_t)(n0 + ty + i) * K + (k0 + tx)] = (__bf16)tile[tx][ty + i];
}

// ---------------------------------------------------------------------------
// RMSNorm over D=1024, writes bf16. One block (256 thr) per row.
// ---------------------------------------------------------------------------
__global__ void k_rmsnorm_bf16(const float* __restrict__ x, const float* __restrict__ w,
                               __bf16* __restrict__ out) {
  int row = blockIdx.x;
  int tid = threadIdx.x;                              // 256 threads, 4 elems each
  const float4 xv = ((const float4*)(x + (size_t)row * D_))[tid];
  float ss = xv.x * xv.x + xv.y * xv.y + xv.z * xv.z + xv.w * xv.w;
  #pragma unroll
  for (int o = 16; o > 0; o >>= 1) ss += __shfl_down(ss, o, 32);
  __shared__ float red[8];
  if ((tid & 31) == 0) red[tid >> 5] = ss;
  __syncthreads();
  float tot = 0.f;
  #pragma unroll
  for (int i = 0; i < 8; i++) tot += red[i];
  float r = rsqrtf(tot * (1.0f / (float)D_) + 1e-6f);
  const float4 wv = ((const float4*)w)[tid];
  union { __bf16 b[4]; uint2 u; } pk;
  pk.b[0] = (__bf16)(xv.x * r * wv.x);
  pk.b[1] = (__bf16)(xv.y * r * wv.y);
  pk.b[2] = (__bf16)(xv.z * r * wv.z);
  pk.b[3] = (__bf16)(xv.w * r * wv.w);
  ((uint2*)(out + (size_t)row * D_))[tid] = pk.u;
}

// ---------------------------------------------------------------------------
// WMMA bf16 GEMM: out[M,N] = act(A[M,K] @ W[K,N]), W given transposed WT[N,K].
// Wave tile 64x32 (4 A-frags x 2 B-frags -> 8 WMMAs / K-step, ~21 FLOP/B).
// Block = 8 waves, 4(M) x 2(N) -> 256x64 block tile.
// ACT: 0 = raw f32, 1 = silu f32, 2 = silu bf16
// ---------------------------------------------------------------------------
__device__ __forceinline__ v16bf load_frag(const __bf16* p) {
  union { uint4 u[2]; v16bf v; } f;
  f.u[0] = *(const uint4*)(p);
  f.u[1] = *(const uint4*)(p + 16);
  return f.v;
}

template <int ACT>
__global__ void __launch_bounds__(256)
k_gemm_wmma(const __bf16* __restrict__ A, const __bf16* __restrict__ WT,
            void* __restrict__ out, int M, int N, int K) {
  int lane = threadIdx.x;                 // 0..31
  int wave = threadIdx.y;                 // 0..7
  int wm = wave >> 1, wn = wave & 1;
  int m0 = blockIdx.y * 256 + wm * 64;
  int n0 = blockIdx.x * 64  + wn * 32;
  int r  = lane & 15;
  int kh = (lane >> 4) * 8;
  const __bf16* a0 = A  + (size_t)(m0 + r) * K + kh;
  const __bf16* b0 = WT + (size_t)(n0 + r) * K + kh;
  const size_t aK = (size_t)16 * K;

  v8f acc[4][2] = {};
  for (int kk = 0; kk < K; kk += 32) {
    v16bf af[4], bf[2];
    #pragma unroll
    for (int i = 0; i < 4; i++) af[i] = load_frag(a0 + (size_t)i * aK + kk);
    #pragma unroll
    for (int j = 0; j < 2; j++) bf[j] = load_frag(b0 + (size_t)j * aK + kk);
    __builtin_prefetch(a0 + kk + 128, 0, 3);          // global_prefetch
    __builtin_prefetch(b0 + kk + 128, 0, 3);
    #pragma unroll
    for (int i = 0; i < 4; i++)
      #pragma unroll
      for (int j = 0; j < 2; j++)
        acc[i][j] = __builtin_amdgcn_wmma_f32_16x16x32_bf16(
            false, af[i], false, bf[j], (short)0, acc[i][j], false, false);
  }
  // C/D layout: VGPR e -> M = e + 8*(lane>=16); N = lane&15
  int nc = lane & 15;
  int mo = (lane >> 4) << 3;
  #pragma unroll
  for (int i = 0; i < 4; i++) {
    #pragma unroll
    for (int j = 0; j < 2; j++) {
      #pragma unroll
      for (int e = 0; e < 8; e++) {
        float v = acc[i][j][e];
        if (ACT >= 1) v = v / (1.0f + __expf(-v));    // SiLU
        size_t idx = (size_t)(m0 + i * 16 + mo + e) * N + (n0 + j * 16 + nc);
        if (ACT == 2) ((__bf16*)out)[idx] = (__bf16)v;
        else          ((float*)out)[idx]  = v;
      }
    }
  }
}

// ---------------------------------------------------------------------------
// beta / g heads:  s = h @ W[D,H];  beta = sigmoid, g = -exp(A_log)*softplus(s+dt)
// ---------------------------------------------------------------------------
__global__ void k_betag(const __bf16* __restrict__ hb, const float* __restrict__ bw,
                        const float* __restrict__ aw, const float* __restrict__ A_log,
                        const float* __restrict__ dt_bias,
                        float* __restrict__ beta, float* __restrict__ g) {
  int gid = blockIdx.x * blockDim.x + threadIdx.x;    // B*T*H
  int row = gid >> 3, hh = gid & 7;
  const uint4* hr4 = (const uint4*)(hb + (size_t)row * D_);
  float sb = 0.f, sa = 0.f;
  for (int kk = 0; kk < D_; kk += 8) {
    union { uint4 u; __bf16 b[8]; } hv;
    hv.u = hr4[kk >> 3];
    #pragma unroll
    for (int j = 0; j < 8; j++) {
      float h = (float)hv.b[j];
      sb += h * bw[(kk + j) * H_ + hh];
      sa += h * aw[(kk + j) * H_ + hh];
    }
  }
  beta[gid] = 1.0f / (1.0f + __expf(-sb));
  float z  = sa + dt_bias[hh];
  float sp = (z > 20.0f) ? z : __logf(1.0f + __expf(z));
  g[gid] = -__expf(A_log[hh]) * sp;
}

// ---------------------------------------------------------------------------
// L2-norm in place over last dim HD=128; one 128-thread block per row.
// ---------------------------------------------------------------------------
__global__ void k_l2norm(float* __restrict__ p) {
  size_t base = (size_t)blockIdx.x * HD_;
  int tid = threadIdx.x;
  float v = p[base + tid];
  float ss = v * v;
  #pragma unroll
  for (int o = 16; o > 0; o >>= 1) ss += __shfl_down(ss, o, 32);
  __shared__ float red[4];
  if ((tid & 31) == 0) red[tid >> 5] = ss;
  __syncthreads();
  float tot = red[0] + red[1] + red[2] + red[3];
  p[base + tid] = v * rsqrtf(tot + 1e-6f);
}

// ---------------------------------------------------------------------------
// Gated delta-rule scan. One block per (b,h); 512 threads.
// Thread (kb = tid&3, vc = tid>>2) owns S[kb*32 .. kb*32+31][vc] in registers.
// The 4 partials for a column live in ADJACENT LANES of one wave -> the
// k^T.S and q^T.S reductions are two shfl_xor, no LDS, no barrier.
// q/k/v staged into LDS with double-buffered async global->LDS copies
// (ASYNCcnt) issued one timestep ahead; 1 barrier per step.
// ---------------------------------------------------------------------------
__global__ void __launch_bounds__(512)
k_scan(const float* __restrict__ q, const float* __restrict__ k,
       const float* __restrict__ v, const float* __restrict__ beta,
       const float* __restrict__ g, float* __restrict__ o,
       float* __restrict__ Sout) {
  int b = blockIdx.x >> 3, h = blockIdx.x & 7;
  int tid = threadIdx.x;
  int kb = tid & 3, vc = tid >> 2;                    // vc in 0..127
  __shared__ float stg[2][3][HD_];                    // q,k,v double-buffered
  float S[32];
  #pragma unroll
  for (int j = 0; j < 32; j++) S[j] = 0.f;

  const float* srcs[3] = {q, k, v};
  auto rbase = [&](int t) { return ((size_t)(b * T_ + t) * H_ + h) * HD_; };

#if ASYNC_LDS
  auto issue = [&](int t, int buf) {
    if (tid < 96) {
      int arr = tid >> 5, li = (tid & 31) * 4;        // 16B per lane
      async_cp16(srcs[arr] + rbase(t) + li, &stg[buf][arr][li]);
    }
  };
  issue(0, 0);
#endif

  for (int t = 0; t < T_; ++t) {
    int buf = t & 1;
#if ASYNC_LDS
    async_wait0();
    __syncthreads();                                  // stage[buf] visible
    if (t + 1 < T_) issue(t + 1, buf ^ 1);
#else
    if (tid < 384) {
      int arr = tid >> 7, li = tid & 127;
      stg[buf][arr][li] = srcs[arr][rbase(t) + li];
    }
    __syncthreads();
#endif
    const float* sq = stg[buf][0];
    const float* sk = stg[buf][1];
    const float* sv = stg[buf][2];
    size_t sb = (size_t)(b * T_ + t) * H_ + h;
    float dec = __expf(g[sb]);
    float bt  = beta[sb];

    float p = 0.f;
    #pragma unroll
    for (int j = 0; j < 32; j++) p += sk[kb * 32 + j] * S[j];
    p *= dec;
    p += __shfl_xor(p, 1, 32);                        // reduce over kb (same wave)
    p += __shfl_xor(p, 2, 32);
    float dl = bt * (sv[vc] - p);

    float po = 0.f;
    #pragma unroll
    for (int j = 0; j < 32; j++) {
      float s2 = S[j] * dec + sk[kb * 32 + j] * dl;
      S[j] = s2;
      po += sq[kb * 32 + j] * s2;
    }
    po += __shfl_xor(po, 1, 32);
    po += __shfl_xor(po, 2, 32);
    if (kb == 0) o[rbase(t) + vc] = po;
  }
  #pragma unroll
  for (int j = 0; j < 32; j++) {
    int kr = kb * 32 + j;
    Sout[(((size_t)(b * H_ + h)) * HD_ + kr) * HD_ + vc] = S[j];
  }
}

// ---------------------------------------------------------------------------
// Per-head RMSNorm(o) * gate (gate already SiLU'd), write bf16. 128 thr / row.
// ---------------------------------------------------------------------------
__global__ void k_onorm_gate(const float* __restrict__ o, const float* __restrict__ w,
                             const float* __restrict__ gate, __bf16* __restrict__ out) {
  size_t base = (size_t)blockIdx.x * HD_;
  int tid = threadIdx.x;
  float v = o[base + tid];
  float ss = v * v;
  #pragma unroll
  for (int off = 16; off > 0; off >>= 1) ss += __shfl_down(ss, off, 32);
  __shared__ float red[4];
  if ((tid & 31) == 0) red[tid >> 5] = ss;
  __syncthreads();
  float tot = red[0] + red[1] + red[2] + red[3];
  float r = rsqrtf(tot * (1.0f / (float)HD_) + 1e-6f);
  out[base + tid] = (__bf16)(v * r * w[tid] * gate[base + tid]);
}

// ---------------------------------------------------------------------------
// Residual add:  out = x + a * RES_SCALE
// ---------------------------------------------------------------------------
__global__ void k_resid(const float* __restrict__ x, const float* __restrict__ a,
                        float* __restrict__ out) {
  size_t i = (size_t)blockIdx.x * 256 + threadIdx.x;
  out[i] = x[i] + a[i] * RES_SCALE_;
}

// ---------------------------------------------------------------------------
extern "C" void kernel_launch(void* const* d_in, const int* in_sizes, int n_in,
                              void* d_out, int out_size, void* d_ws, size_t ws_size,
                              hipStream_t stream) {
  (void)in_sizes; (void)n_in; (void)out_size; (void)ws_size;
  const float* x      = (const float*)d_in[0];
  const float* ln1_w  = (const float*)d_in[1];
  const float* qw     = (const float*)d_in[2];
  const float* kw     = (const float*)d_in[3];
  const float* vw     = (const float*)d_in[4];
  const float* bw     = (const float*)d_in[5];
  const float* aw     = (const float*)d_in[6];
  const float* A_log  = (const float*)d_in[7];
  const float* dt_b   = (const float*)d_in[8];
  const float* gw     = (const float*)d_in[9];
  const float* onw    = (const float*)d_in[10];
  const float* ow     = (const float*)d_in[11];
  const float* ln2_w  = (const float*)d_in[12];
  const float* fcw    = (const float*)d_in[13];
  const float* pw     = (const float*)d_in[14];

  float* y  = (float*)d_out;                       // [B,T,D]
  float* Sf = y + (size_t)B_ * T_ * D_;            // [B,H,HD,HD]

  const size_t MROWS = (size_t)B_ * T_;            // 8192
  char* ws = (char*)d_ws;
  size_t off = 0;
  auto alloc = [&](size_t bytes) -> char* {
    char* p = ws + off;
    off = (off + bytes + 255) & ~(size_t)255;
    return p;
  };
  __bf16* wtq  = (__bf16*)alloc((size_t)D_ * D_ * 2);
  __bf16* wtk  = (__bf16*)alloc((size_t)D_ * D_ * 2);
  __bf16* wtv  = (__bf16*)alloc((size_t)D_ * D_ * 2);
  __bf16* wtg  = (__bf16*)alloc((size_t)D_ * D_ * 2);
  __bf16* wto  = (__bf16*)alloc((size_t)D_ * D_ * 2);
  __bf16* wtfc = (__bf16*)alloc((size_t)D_ * FOUR_D_ * 2);   // [4096,1024]
  __bf16* wtp  = (__bf16*)alloc((size_t)D_ * FOUR_D_ * 2);   // [1024,4096]
  __bf16* hb   = (__bf16*)alloc(MROWS * D_ * 2);
  float*  qb   = (float*)alloc(MROWS * D_ * 4);
  float*  kbuf = (float*)alloc(MROWS * D_ * 4);
  float*  vb   = (float*)alloc(MROWS * D_ * 4);              // vb+gateb overlaid by mid
  float*  gateb= (float*)alloc(MROWS * D_ * 4);
  float*  betab= (float*)alloc(MROWS * H_ * 4);
  float*  gb   = (float*)alloc(MROWS * H_ * 4);
  float*  obuf = (float*)alloc(MROWS * D_ * 4);
  __bf16* obn  = (__bf16*)alloc(MROWS * D_ * 2);
  // aliases (producers fully consumed before reuse)
  float*  attn = qb;                               // ow output
  float*  x1   = kbuf;                             // residual 1
  __bf16* h2b  = hb;                               // ln2 output
  __bf16* mid  = (__bf16*)vb;                      // [8192,4096] bf16 = vb+gateb span
  float*  mlp  = obuf;                             // pw output

  dim3 tb(32, 8);
  // 1) weight transpose + bf16 convert
  k_transpose_cvt<<<dim3(D_/32, D_/32), tb, 0, stream>>>(qw,  wtq,  D_, D_);
  k_transpose_cvt<<<dim3(D_/32, D_/32), tb, 0, stream>>>(kw,  wtk,  D_, D_);
  k_transpose_cvt<<<dim3(D_/32, D_/32), tb, 0, stream>>>(vw,  wtv,  D_, D_);
  k_transpose_cvt<<<dim3(D_/32, D_/32), tb, 0, stream>>>(gw,  wtg,  D_, D_);
  k_transpose_cvt<<<dim3(D_/32, D_/32), tb, 0, stream>>>(ow,  wto,  D_, D_);
  k_transpose_cvt<<<dim3(FOUR_D_/32, D_/32), tb, 0, stream>>>(fcw, wtfc, D_, FOUR_D_);
  k_transpose_cvt<<<dim3(D_/32, FOUR_D_/32), tb, 0, stream>>>(pw,  wtp,  FOUR_D_, D_);

  // 2) ln1
  k_rmsnorm_bf16<<<(int)MROWS, 256, 0, stream>>>(x, ln1_w, hb);

  // 3) projections (WMMA, 256x64 block tile)
  dim3 gg(D_/64, (int)(MROWS/256));
  k_gemm_wmma<1><<<gg, tb, 0, stream>>>(hb, wtq, qb,    (int)MROWS, D_, D_);
  k_gemm_wmma<1><<<gg, tb, 0, stream>>>(hb, wtk, kbuf,  (int)MROWS, D_, D_);
  k_gemm_wmma<1><<<gg, tb, 0, stream>>>(hb, wtv, vb,    (int)MROWS, D_, D_);
  k_gemm_wmma<1><<<gg, tb, 0, stream>>>(hb, wtg, gateb, (int)MROWS, D_, D_);

  // 4) l2norm q,k  +  beta/g heads
  k_l2norm<<<(int)(MROWS * H_), HD_, 0, stream>>>(qb);
  k_l2norm<<<(int)(MROWS * H_), HD_, 0, stream>>>(kbuf);
  k_betag<<<(int)(MROWS * H_ / 256), 256, 0, stream>>>(hb, bw, aw, A_log, dt_b, betab, gb);

  // 5) recurrent scan (writes o and final state)
  k_scan<<<B_ * H_, 512, 0, stream>>>(qb, kbuf, vb, betab, gb, obuf, Sf);

  // 6) per-head norm * gate -> bf16
  k_onorm_gate<<<(int)(MROWS * H_), HD_, 0, stream>>>(obuf, onw, gateb, obn);

  // 7) output projection + residual
  k_gemm_wmma<0><<<gg, tb, 0, stream>>>(obn, wto, attn, (int)MROWS, D_, D_);
  k_resid<<<(int)(MROWS * D_ / 256), 256, 0, stream>>>(x, attn, x1);

  // 8) MLP
  k_rmsnorm_bf16<<<(int)MROWS, 256, 0, stream>>>(x1, ln2_w, h2b);
  k_gemm_wmma<2><<<dim3(FOUR_D_/64, (int)(MROWS/256)), tb, 0, stream>>>(h2b, wtfc, mid, (int)MROWS, FOUR_D_, D_);
  k_gemm_wmma<0><<<dim3(D_/64, (int)(MROWS/256)),      tb, 0, stream>>>(mid, wtp,  mlp, (int)MROWS, D_, FOUR_D_);
  k_resid<<<(int)(MROWS * D_ / 256), 256, 0, stream>>>(x1, mlp, y);
}
